// GNNLayer_63488206570152
// MI455X (gfx1250) — matrix operand order repaired and checked by reading, not compiled, not run on hardware
//
#include <hip/hip_runtime.h>
#include <math.h>

#define H 128
#define BGR 64

typedef __bf16 bf16_t;
typedef __attribute__((ext_vector_type(16))) __bf16 v16bf;
typedef __attribute__((ext_vector_type(8)))  __bf16 v8bf;
typedef __attribute__((ext_vector_type(8)))  float  v8f;

// ---------------------------------------------------------------------------
// A-fragment (16x32 bf16, wave32): lane L<16 -> row m0+L, K = kh..kh+7 and
// kh+16..kh+23 with kh=0; lanes 16..31 -> same rows, kh=8.
// p points at f32 row_base + kt*32 + kh.
// ---------------------------------------------------------------------------
__device__ __forceinline__ v16bf make_afrag(const float* __restrict__ p) {
  v16bf a;
#pragma unroll
  for (int j = 0; j < 8; ++j) {
    a[j]     = (bf16_t)p[j];
    a[j + 8] = (bf16_t)p[j + 16];
  }
  return a;
}

// Same fragment but from a bf16 row (LDS tile): two raw 16B loads, no cvt.
__device__ __forceinline__ v16bf make_afrag_bf16(const bf16_t* p) {
  const v8bf lo = *(const v8bf*)(p);
  const v8bf hi = *(const v8bf*)(p + 16);
  v16bf a;
#pragma unroll
  for (int j = 0; j < 8; ++j) {
    a[j]     = lo[j];
    a[j + 8] = hi[j];
  }
  return a;
}

// ---------------------------------------------------------------------------
// GEMM of one 16-row tile against a pre-packed [KT][8][32 lanes][16] bf16
// weight image (global or LDS) -> 16x128 f32 accumulator (8 N-tiles).
// ---------------------------------------------------------------------------
template <int KT>
__device__ __forceinline__ void wmma_gemm(const v16bf* __restrict__ afr,
                                          const bf16_t* pack,
                                          v8f* __restrict__ acc, int lane) {
#pragma unroll
  for (int kt = 0; kt < KT; ++kt) {
#pragma unroll
    for (int nt = 0; nt < 8; ++nt) {
      const v16bf b = *(const v16bf*)(pack + ((((kt << 3) + nt) << 5) + lane) * 16);
      acc[nt] = __builtin_amdgcn_wmma_f32_16x16x32_bf16(
          false, afr[kt], false, b, (short)0, acc[nt], false, false);
    }
  }
}

// ---------------------------------------------------------------------------
// Pack a row-major f32 [K x 128] matrix into WMMA-B fragment order (bf16).
// ---------------------------------------------------------------------------
__global__ void pack_b_kernel(const float* __restrict__ W, bf16_t* __restrict__ out, int KT) {
  const int total = KT << 12;  // KT * 4096
  for (int o = blockIdx.x * blockDim.x + threadIdx.x; o < total;
       o += gridDim.x * blockDim.x) {
    const int j    = o & 15;
    const int lane = (o >> 4) & 31;
    const int ntk  = o >> 9;
    const int nt   = ntk & 7;
    const int kt   = ntk >> 3;
    const int n    = (nt << 4) + (lane & 15);
    const int k    = (kt << 5) + ((lane >> 4) << 3) + (j & 7) + ((j >> 3) << 4);
    out[o] = (bf16_t)W[k * H + n];
  }
}

// ---------------------------------------------------------------------------
// Fused NNConv edge pass: gate = relu(relu(hE@w1+b1)@w2+b2);
// agg[dst] += hin[src] * gate   (f32 atomics into L2-resident agg)
// w1 fragments: global (register/L0 resident under 512-VGPR cap);
// w2 fragments: staged in LDS, ds_load_b128-fed.
// ---------------------------------------------------------------------------
__global__ void __launch_bounds__(128, 2)
edge_msg_kernel(const float* __restrict__ hE, const float* __restrict__ hin,
                const int* __restrict__ src, const int* __restrict__ dst,
                const bf16_t* __restrict__ w1p, const bf16_t* __restrict__ w2p,
                const float* __restrict__ b1, const float* __restrict__ b2,
                float* __restrict__ agg, int nE) {
  alignas(32) __shared__ bf16_t w2s[16384];        // 32 KB packed w2
  alignas(32) __shared__ bf16_t tile[4][16 * H];   // 4 KB/wave t1 (bf16)
  const int lane = threadIdx.x & 31;
  const int wv   = threadIdx.x >> 5;
  const int gw   = gridDim.x * 4;
  const int cl   = lane & 15;
  const int rh   = (lane >> 4) * 8;
  bf16_t* tl = tile[wv];

  {  // cooperative LDS fill of packed w2 (2048 x uint4)
    const uint4* sw = (const uint4*)w2p;
    uint4* dw = (uint4*)w2s;
    for (int i = threadIdx.x; i < 2048; i += 128) dw[i] = sw[i];
  }
  __syncthreads();

  float bias1[8], bias2[8];
#pragma unroll
  for (int nt = 0; nt < 8; ++nt) {
    bias1[nt] = b1[nt * 16 + cl];
    bias2[nt] = b2[nt * 16 + cl];
  }

  const int ntiles = nE >> 4;
  for (int t = blockIdx.x * 4 + wv; t < ntiles; t += gw) {
    const int m0 = t << 4;
    if (t + gw < ntiles)
      __builtin_prefetch(hE + (size_t)(t + gw) * 16 * H, 0, 1);  // global_prefetch_b8

    // --- GEMM1: hE tile @ w1 ---
    const float* rp = hE + (size_t)(m0 + cl) * H + rh;
    v16bf afr[4];
#pragma unroll
    for (int kt = 0; kt < 4; ++kt) afr[kt] = make_afrag(rp + kt * 32);
    v8f acc[8] = {};
    wmma_gemm<4>(afr, w1p, acc, lane);

    // bias + relu -> bf16 LDS tile, row-major [16][128]
#pragma unroll
    for (int nt = 0; nt < 8; ++nt)
#pragma unroll
      for (int v = 0; v < 8; ++v) {
        float x = acc[nt][v] + bias1[nt];
        tl[(rh + v) * H + nt * 16 + cl] = (bf16_t)(x > 0.f ? x : 0.f);
      }

    // --- GEMM2: t1 @ w2 (A from LDS transpose, B from LDS) ---
    const bf16_t* lp = tl + cl * H + rh;
#pragma unroll
    for (int kt = 0; kt < 4; ++kt) afr[kt] = make_afrag_bf16(lp + kt * 32);
    v8f acc2[8] = {};
    wmma_gemm<4>(afr, w2s, acc2, lane);

    // --- msg = hin[src] * gate ; atomic scatter into agg[dst] ---
    const int sl = src[m0 + cl];   // 16 distinct values, broadcast via shfl
    const int dl = dst[m0 + cl];
#pragma unroll
    for (int v = 0; v < 8; ++v) {
      const int r = rh + v;
      const int s = __shfl(sl, r, 32);
      const int d = __shfl(dl, r, 32);
      const float* hs = hin + (size_t)s * H;
      float* ag = agg + (size_t)d * H;
#pragma unroll
      for (int nt = 0; nt < 8; ++nt) {
        float g = acc2[nt][v] + bias2[nt];
        g = g > 0.f ? g : 0.f;
        const int c = nt * 16 + cl;
        atomicAdd(&ag[c], hs[c] * g);
      }
    }
  }
}

// ---------------------------------------------------------------------------
// Fused node pass: out = LayerNorm(resid + hin@root + agg + cbias)
// ---------------------------------------------------------------------------
__global__ void __launch_bounds__(128, 2)
node_kernel(const float* __restrict__ hin, const float* __restrict__ resid,
            const float* __restrict__ agg, const bf16_t* __restrict__ rootp,
            const float* __restrict__ cbias, const float* __restrict__ lg,
            const float* __restrict__ lb, float* __restrict__ hout, int nN) {
  __shared__ float tile[4][16 * H];
  const int lane = threadIdx.x & 31;
  const int wv   = threadIdx.x >> 5;
  const int gw   = gridDim.x * 4;
  const int cl   = lane & 15;
  const int rh   = (lane >> 4) * 8;
  float* tl = tile[wv];

  const int ntiles = nN >> 4;
  for (int t = blockIdx.x * 4 + wv; t < ntiles; t += gw) {
    const int m0 = t << 4;
    const float* rp = hin + (size_t)(m0 + cl) * H + rh;
    v16bf afr[4];
#pragma unroll
    for (int kt = 0; kt < 4; ++kt) afr[kt] = make_afrag(rp + kt * 32);
    v8f acc[8] = {};
    wmma_gemm<4>(afr, rootp, acc, lane);

#pragma unroll
    for (int nt = 0; nt < 8; ++nt)
#pragma unroll
      for (int v = 0; v < 8; ++v)
        tl[(rh + v) * H + nt * 16 + cl] = acc[nt][v];

    // coalesced residual + aggregation + bias add
#pragma unroll 4
    for (int i = 0; i < 64; ++i) {
      const int idx = i * 32 + lane;
      const int r = idx >> 7, c = idx & (H - 1);
      const size_t g = (size_t)(m0 + r) * H + c;
      tl[idx] += resid[g] + agg[g] + cbias[c];
    }

    // LayerNorm: 2 lanes per row, __shfl_xor combine (wave32)
    const int r  = lane >> 1;
    const int hf = lane & 1;
    const float* q = tl + r * H + hf * 64;
    float s1 = 0.f, s2 = 0.f;
#pragma unroll 8
    for (int j = 0; j < 64; ++j) { const float x = q[j]; s1 += x; s2 += x * x; }
    s1 += __shfl_xor(s1, 1, 32);
    s2 += __shfl_xor(s2, 1, 32);
    const float mu  = s1 * (1.f / H);
    const float inv = rsqrtf(s2 * (1.f / H) - mu * mu + 1e-5f);
    float* op = hout + (size_t)(m0 + r) * H + hf * 64;
    const float* gp = lg + hf * 64;
    const float* bp = lb + hf * 64;
#pragma unroll 8
    for (int j = 0; j < 64; ++j) op[j] = (q[j] - mu) * inv * gp[j] + bp[j];
  }
}

// ---------------------------------------------------------------------------
// EdgeMLP: e = hE + (gelu([h_src,hE,h_dst]@W11+b11)@W12+b12);
// W11 streamed from global/L0 (too large to hoist), W12 in LDS.
// Epilogue writes straight from the C layout + per-lane BN column partials.
// ---------------------------------------------------------------------------
__global__ void __launch_bounds__(128, 2)
edge_mlp_kernel(const float* __restrict__ hE, const float* __restrict__ hN,
                const int* __restrict__ src, const int* __restrict__ dst,
                const bf16_t* __restrict__ W11p, const float* __restrict__ b11,
                const bf16_t* __restrict__ W12p, const float* __restrict__ b12,
                float* __restrict__ eout, float* __restrict__ colsum,
                float* __restrict__ colsq, int nE) {
  alignas(32) __shared__ bf16_t w12s[16384];       // 32 KB packed W12
  alignas(32) __shared__ bf16_t tile[4][16 * H];   // 4 KB/wave gelu(t1) bf16
  const int lane = threadIdx.x & 31;
  const int wv   = threadIdx.x >> 5;
  const int gw   = gridDim.x * 4;
  const int cl   = lane & 15;
  const int rh   = (lane >> 4) * 8;
  bf16_t* tl = tile[wv];

  {
    const uint4* sw = (const uint4*)W12p;
    uint4* dw = (uint4*)w12s;
    for (int i = threadIdx.x; i < 2048; i += 128) dw[i] = sw[i];
  }
  __syncthreads();

  float bias1[8], bias2[8];
#pragma unroll
  for (int nt = 0; nt < 8; ++nt) {
    bias1[nt] = b11[nt * 16 + cl];
    bias2[nt] = b12[nt * 16 + cl];
  }
  float ls[8] = {}, lq[8] = {};

  const int ntiles = nE >> 4;
  for (int t = blockIdx.x * 4 + wv; t < ntiles; t += gw) {
    const int m0 = t << 4;
    const int s  = src[m0 + cl];
    const int d  = dst[m0 + cl];
    const float* ps = hN + (size_t)s * H + rh;
    const float* pe = hE + (size_t)(m0 + cl) * H + rh;
    const float* pd = hN + (size_t)d * H + rh;

    v16bf afr[12];
#pragma unroll
    for (int kt = 0; kt < 4; ++kt) afr[kt]     = make_afrag(ps + kt * 32);
#pragma unroll
    for (int kt = 0; kt < 4; ++kt) afr[4 + kt] = make_afrag(pe + kt * 32);
#pragma unroll
    for (int kt = 0; kt < 4; ++kt) afr[8 + kt] = make_afrag(pd + kt * 32);

    v8f acc[8] = {};
    wmma_gemm<12>(afr, W11p, acc, lane);

    // bias + exact-erf GELU -> bf16 LDS tile
#pragma unroll
    for (int nt = 0; nt < 8; ++nt)
#pragma unroll
      for (int v = 0; v < 8; ++v) {
        float x = acc[nt][v] + bias1[nt];
        x = 0.5f * x * (1.f + erff(x * 0.70710678118654752f));
        tl[(rh + v) * H + nt * 16 + cl] = (bf16_t)x;
      }

    const bf16_t* lp = tl + cl * H + rh;
    v16bf a2[4];
#pragma unroll
    for (int kt = 0; kt < 4; ++kt) a2[kt] = make_afrag_bf16(lp + kt * 32);
    v8f acc2[8] = {};
    wmma_gemm<4>(a2, w12s, acc2, lane);

    // residual add + store from C layout; per-lane BN column partials
#pragma unroll
    for (int v = 0; v < 8; ++v) {
      const size_t rowg = (size_t)(m0 + rh + v) * H;
#pragma unroll
      for (int nt = 0; nt < 8; ++nt) {
        const int c = nt * 16 + cl;
        const float e = acc2[nt][v] + bias2[nt] + hE[rowg + c];
        eout[rowg + c] = e;
        ls[nt] += e;
        lq[nt] += e * e;
      }
    }
  }
#pragma unroll
  for (int nt = 0; nt < 8; ++nt) {
    atomicAdd(&colsum[nt * 16 + cl], ls[nt]);
    atomicAdd(&colsq[nt * 16 + cl], lq[nt]);
  }
}

__global__ void bn_finalize_kernel(const float* __restrict__ colsum,
                                   const float* __restrict__ colsq,
                                   const float* __restrict__ bng,
                                   const float* __restrict__ bnb,
                                   float* __restrict__ scale,
                                   float* __restrict__ shift, float invE) {
  const int c = threadIdx.x;
  const float mu  = colsum[c] * invE;
  const float var = colsq[c] * invE - mu * mu;
  const float sc  = bng[c] * rsqrtf(var + 1e-5f);
  scale[c] = sc;
  shift[c] = bnb[c] - mu * sc;
}

__global__ void bn_apply_kernel(float* __restrict__ e, const float* __restrict__ sc,
                                const float* __restrict__ sh, size_t n4) {
  const size_t t = (size_t)blockIdx.x * blockDim.x + threadIdx.x;
  if (t >= n4) return;
  float4 v = reinterpret_cast<float4*>(e)[t];
  const int c = (int)((t << 2) & (H - 1));
  v.x = v.x * sc[c]     + sh[c];
  v.y = v.y * sc[c + 1] + sh[c + 1];
  v.z = v.z * sc[c + 2] + sh[c + 2];
  v.w = v.w * sc[c + 3] + sh[c + 3];
  reinterpret_cast<float4*>(e)[t] = v;
}

__global__ void ctx_count_kernel(const int* __restrict__ batch, float* __restrict__ cnt, int nN) {
  const int t = blockIdx.x * blockDim.x + threadIdx.x;
  if (t < nN) atomicAdd(&cnt[batch[t]], 1.0f);
}

__global__ void ctx_accum_kernel(const float* __restrict__ h, const int* __restrict__ batch,
                                 float* __restrict__ csum, int nN) {
  const int t = blockIdx.x * blockDim.x + threadIdx.x;
  const int n = t >> 5;
  if (n >= nN) return;
  const int c0 = (t & 31) << 2;
  const int b = batch[n];
  const float* hp = h + (size_t)n * H + c0;
  float* cp = csum + (size_t)b * H + c0;
#pragma unroll
  for (int j = 0; j < 4; ++j) atomicAdd(&cp[j], hp[j]);
}

__global__ void ctx_gate_kernel(const float* __restrict__ csum, const float* __restrict__ cnt,
                                const float* __restrict__ w1, const float* __restrict__ bb1,
                                const float* __restrict__ w2, const float* __restrict__ bb2,
                                float* __restrict__ gateB) {
  __shared__ float cV[H];
  __shared__ float t1[H];
  const int b = blockIdx.x;
  const int c = threadIdx.x;
  cV[c] = csum[(size_t)b * H + c] / fmaxf(cnt[b], 1.0f);
  __syncthreads();
  float s = bb1[c];
#pragma unroll 8
  for (int k = 0; k < H; ++k) s += cV[k] * w1[k * H + c];
  t1[c] = s > 0.f ? s : 0.f;
  __syncthreads();
  float u = bb2[c];
#pragma unroll 8
  for (int k = 0; k < H; ++k) u += t1[k] * w2[k * H + c];
  gateB[(size_t)b * H + c] = 1.f / (1.f + expf(-u));
}

__global__ void gating_apply_kernel(const float* __restrict__ h, const int* __restrict__ batch,
                                    const float* __restrict__ gateB, float* __restrict__ outV,
                                    int nN) {
  const int t = blockIdx.x * blockDim.x + threadIdx.x;
  const int n = t >> 5;
  if (n >= nN) return;
  const int c0 = (t & 31) << 2;
  const int b = batch[n];
  const float* hp = h + (size_t)n * H + c0;
  const float* gp = gateB + (size_t)b * H + c0;
  float4 v;
  v.x = hp[0] * gp[0];
  v.y = hp[1] * gp[1];
  v.z = hp[2] * gp[2];
  v.w = hp[3] * gp[3];
  *reinterpret_cast<float4*>(outV + (size_t)n * H + c0) = v;
}

// ---------------------------------------------------------------------------
extern "C" void kernel_launch(void* const* d_in, const int* in_sizes, int n_in,
                              void* d_out, int out_size, void* d_ws, size_t ws_size,
                              hipStream_t stream) {
  (void)n_in; (void)out_size; (void)ws_size;
  const float* h_V       = (const float*)d_in[0];
  const int*   eidx      = (const int*)d_in[1];
  const float* hE        = (const float*)d_in[2];
  const int*   batch     = (const int*)d_in[3];
  const float* conv_w1   = (const float*)d_in[4];
  const float* conv_b1   = (const float*)d_in[5];
  const float* conv_w2   = (const float*)d_in[6];
  const float* conv_b2   = (const float*)d_in[7];
  const float* conv_root = (const float*)d_in[8];
  const float* conv_bias = (const float*)d_in[9];
  const float* ln0g      = (const float*)d_in[10];
  const float* ln0b      = (const float*)d_in[11];
  const float* ln1g      = (const float*)d_in[12];
  const float* ln1b      = (const float*)d_in[13];
  const float* W11       = (const float*)d_in[14];
  const float* b11       = (const float*)d_in[15];
  const float* W12       = (const float*)d_in[16];
  const float* b12       = (const float*)d_in[17];
  const float* bng       = (const float*)d_in[18];
  const float* bnb       = (const float*)d_in[19];
  const float* cw1       = (const float*)d_in[20];
  const float* cb1       = (const float*)d_in[21];
  const float* cw2       = (const float*)d_in[22];
  const float* cb2       = (const float*)d_in[23];

  const int N = in_sizes[0] / H;
  const int E = in_sizes[2] / H;
  const int* src = eidx;
  const int* dst = eidx + E;

  // ---- workspace carve-up (256B aligned) ----
  char* ws = (char*)d_ws;
  size_t off = 0;
  auto take = [&](size_t bytes) -> char* {
    char* p = ws + off;
    off = (off + bytes + 255) & ~(size_t)255;
    return p;
  };
  bf16_t* w1p   = (bf16_t*)take(4  * 4096 * sizeof(bf16_t));
  bf16_t* w2p   = (bf16_t*)take(4  * 4096 * sizeof(bf16_t));
  bf16_t* rootp = (bf16_t*)take(4  * 4096 * sizeof(bf16_t));
  bf16_t* W11p  = (bf16_t*)take(12 * 4096 * sizeof(bf16_t));
  bf16_t* W12p  = (bf16_t*)take(4  * 4096 * sizeof(bf16_t));
  float* colsum  = (float*)take(H * sizeof(float));
  float* colsq   = (float*)take(H * sizeof(float));
  float* bnscale = (float*)take(H * sizeof(float));
  float* bnshift = (float*)take(H * sizeof(float));
  float* cnt     = (float*)take(BGR * sizeof(float));
  float* csum    = (float*)take((size_t)BGR * H * sizeof(float));
  float* gateB   = (float*)take((size_t)BGR * H * sizeof(float));
  float* h1      = (float*)take((size_t)N * H * sizeof(float));
  float* h2      = (float*)take((size_t)N * H * sizeof(float));
  float* agg     = (float*)take((size_t)N * H * sizeof(float));

  float* outV = (float*)d_out;
  float* outE = outV + (size_t)N * H;

  // ---- pack weights into WMMA B-fragment order (bf16) ----
  pack_b_kernel<<<64, 256, 0, stream>>>(conv_w1, w1p, 4);
  pack_b_kernel<<<64, 256, 0, stream>>>(conv_w2, w2p, 4);
  pack_b_kernel<<<64, 256, 0, stream>>>(conv_root, rootp, 4);
  pack_b_kernel<<<192, 256, 0, stream>>>(W11, W11p, 12);
  pack_b_kernel<<<64, 256, 0, stream>>>(W12, W12p, 4);

  // ---- NNConv pass 1 ----
  hipMemsetAsync(agg, 0, (size_t)N * H * sizeof(float), stream);
  edge_msg_kernel<<<1280, 128, 0, stream>>>(hE, h_V, src, dst, w1p, w2p,
                                            conv_b1, conv_b2, agg, E);
  node_kernel<<<320, 128, 0, stream>>>(h_V, h_V, agg, rootp, conv_bias,
                                       ln0g, ln0b, h1, N);

  // ---- NNConv pass 2 (gate recomputed; residual is original h_V) ----
  hipMemsetAsync(agg, 0, (size_t)N * H * sizeof(float), stream);
  edge_msg_kernel<<<1280, 128, 0, stream>>>(hE, h1, src, dst, w1p, w2p,
                                            conv_b1, conv_b2, agg, E);
  node_kernel<<<320, 128, 0, stream>>>(h1, h_V, agg, rootp, conv_bias,
                                       ln1g, ln1b, h2, N);

  // ---- EdgeMLP + BatchNorm stats ----
  hipMemsetAsync(colsum, 0, H * sizeof(float), stream);
  hipMemsetAsync(colsq, 0, H * sizeof(float), stream);
  hipMemsetAsync(cnt, 0, BGR * sizeof(float), stream);
  hipMemsetAsync(csum, 0, (size_t)BGR * H * sizeof(float), stream);

  edge_mlp_kernel<<<1280, 128, 0, stream>>>(hE, h2, src, dst, W11p, b11, W12p,
                                            b12, outE, colsum, colsq, E);
  bn_finalize_kernel<<<1, H, 0, stream>>>(colsum, colsq, bng, bnb, bnscale,
                                          bnshift, 1.0f / (float)E);
  const size_t n4 = (size_t)E * H / 4;
  bn_apply_kernel<<<(unsigned)((n4 + 255) / 256), 256, 0, stream>>>(outE, bnscale,
                                                                    bnshift, n4);

  // ---- Context gating ----
  ctx_count_kernel<<<(N + 255) / 256, 256, 0, stream>>>(batch, cnt, N);
  ctx_accum_kernel<<<(N * 32 + 255) / 256, 256, 0, stream>>>(h2, batch, csum, N);
  ctx_gate_kernel<<<BGR, H, 0, stream>>>(csum, cnt, cw1, cb1, cw2, cb2, gateB);
  gating_apply_kernel<<<(N * 32 + 255) / 256, 256, 0, stream>>>(h2, batch, gateB,
                                                                outV, N);
}
